// GraphAttentionLayer_30253749633497
// MI455X (gfx1250) — compile-verified, matrix operand order checked by reading, not run on hardware
//
#include <hip/hip_runtime.h>
#include <hip/hip_bf16.h>

typedef __attribute__((ext_vector_type(16))) _Float16 v16h;
typedef __attribute__((ext_vector_type(8)))  float    v8f;
typedef __attribute__((ext_vector_type(4)))  float    v4f;

#define B_ 8
#define N_ 1024
#define F_ 128
#define E_ 16
#define W_ 8                      // waves per block in attn kernel (j-split factor)

// K index inside a 16-bit WMMA A/B fragment for element e (0..15) of a lane.
// ISA 7.12.2: lanes 0-15 hold K {0..7, 16..23}; lanes 16-31 hold K {8..15, 24..31};
// element e = 2*vgpr + halfsel maps to consecutive K within each 8-group.
__device__ __forceinline__ int frag_k(int e, int hi) {
    return ((e < 8) ? e : e + 8) + (hi ? 8 : 0);
}

// ---------------------------------------------------------------------------
// Pack Wa [128,128] f32 (row-major, K x N) into f16 B-fragment layout:
// halves index = (((kt*8 + ft)*32 + lane)*16 + e)
// ---------------------------------------------------------------------------
__global__ void pack_wa_kernel(const float* __restrict__ Wa, _Float16* __restrict__ waPack) {
    int tid  = blockIdx.x * blockDim.x + threadIdx.x;
    int wave = tid >> 5, lane = tid & 31, hi = lane >> 4, col = lane & 15;
    int kt = wave >> 3, ft = wave & 7;              // 4 k-tiles x 8 f-tiles = 32 waves
    _Float16* dst = waPack + ((size_t)((kt * 8 + ft) * 32 + lane)) * 16;
#pragma unroll
    for (int e = 0; e < 16; ++e) {
        int k = kt * 32 + frag_k(e, hi);
        dst[e] = (_Float16)Wa[(size_t)k * F_ + ft * 16 + col];
    }
}

// ---------------------------------------------------------------------------
// Wh = h @ Wa : each wave computes a 16-row x 128-col tile with WMMA f16.
// 512 waves total (8192 rows / 16).
// ---------------------------------------------------------------------------
__global__ void wh_kernel(const float* __restrict__ h, const _Float16* __restrict__ waPack,
                          float* __restrict__ Wh) {
    int tid  = blockIdx.x * blockDim.x + threadIdx.x;
    int wave = tid >> 5, lane = tid & 31, hi = lane >> 4, col = lane & 15;
    int row0 = wave * 16;
    const float* hrow = h + (size_t)(row0 + col) * F_;   // this lane's A-fragment row
    int klo = hi ? 8 : 0;

    v8f acc[8] = {};
#pragma unroll
    for (int kt = 0; kt < 4; ++kt) {
        v16h a;
        const float* p0 = hrow + kt * 32 + klo;          // K = kt*32 + klo + 0..7
        const float* p1 = hrow + kt * 32 + 16 + klo;     // K = kt*32 + 16 + klo + 0..7
#pragma unroll
        for (int e = 0; e < 8; ++e) a[e] = (_Float16)p0[e];
#pragma unroll
        for (int e = 0; e < 8; ++e) a[8 + e] = (_Float16)p1[e];
#pragma unroll
        for (int ft = 0; ft < 8; ++ft) {
            v16h b = *((const v16h*)waPack + ((kt * 8 + ft) * 32 + lane));
            acc[ft] = __builtin_amdgcn_wmma_f32_16x16x32_f16(
                false, a, false, b, (short)0, acc[ft], false, false);
        }
    }
    int base = hi ? 8 : 0;                               // C/D rows M = r + base, col = ft*16+col
#pragma unroll
    for (int ft = 0; ft < 8; ++ft)
#pragma unroll
        for (int r = 0; r < 8; ++r)
            Wh[(size_t)(row0 + base + r) * F_ + ft * 16 + col] = acc[ft][r];
}

// ---------------------------------------------------------------------------
// s = Wh @ a_src, d = Wh @ a_dst : one wave per row (8192 waves).
// ---------------------------------------------------------------------------
__global__ void sd_kernel(const float* __restrict__ Wh, const float* __restrict__ a_src,
                          const float* __restrict__ a_dst,
                          float* __restrict__ s, float* __restrict__ d) {
    int tid  = blockIdx.x * blockDim.x + threadIdx.x;
    int wave = tid >> 5, lane = tid & 31;
    const float* w = Wh + (size_t)wave * F_;
    float ps = 0.f, pd = 0.f;
#pragma unroll
    for (int f = lane; f < F_; f += 32) {
        float v = w[f];
        ps += v * a_src[f];
        pd += v * a_dst[f];
    }
#pragma unroll
    for (int m = 16; m >= 1; m >>= 1) {
        ps += __shfl_xor(ps, m, 32);
        pd += __shfl_xor(pd, m, 32);
    }
    if (lane == 0) { s[wave] = ps; d[wave] = pd; }
}

// ---------------------------------------------------------------------------
// Pack Wh (f32 row-major per batch: [N_ nodes x F_]) into f16 B-fragment layout
// for the attention GEMM (K = node index j): 2048 waves.
// halves index = ((((b*32 + kt)*8 + ft)*32 + lane)*16 + e)
// ---------------------------------------------------------------------------
__global__ void pack_wh_kernel(const float* __restrict__ Wh, _Float16* __restrict__ whPack) {
    int tid  = blockIdx.x * blockDim.x + threadIdx.x;
    int wave = tid >> 5, lane = tid & 31, hi = lane >> 4, col = lane & 15;
    int kt = (wave >> 3) & 31;     // 32 j-tiles of 32
    int ft = wave & 7;             // 8 f-tiles of 16
    int b  = wave >> 8;            // 8 batches
    _Float16* dst = whPack + ((size_t)wave * 32 + lane) * 16;
#pragma unroll
    for (int e = 0; e < 16; ++e) {
        int j = kt * 32 + frag_k(e, hi);
        dst[e] = (_Float16)Wh[(size_t)(b * N_ + j) * F_ + ft * 16 + col];
    }
}

// ---------------------------------------------------------------------------
// Fused attention. One block (8 waves) per 16-row i-tile; wave w handles
// j-tiles kt = w, w+8, w+16, w+24 (j split 8 ways => 4096 waves total, enough
// outstanding NT loads to approach 23.3 TB/s on the 512 MB edge stream).
//   per kt: preload 8 Wh B-fragments (registers), then each lane computes the
//   16 scores its A-fragment slots need: t = tanh(s_i + d_j + edges·a_edge)
//   (t in (-1,1) => exp never overflows, no max pass), 8x WMMA accumulate.
// Partials combined across waves via ds_add_f32 into an 8 KB LDS tile, then
// normalized by softmax row sums, final tanh, coalesced stores.
// ---------------------------------------------------------------------------
__global__ void __launch_bounds__(32 * W_)
attn_kernel(const float* __restrict__ edges, const float* __restrict__ a_edge,
            const float* __restrict__ s, const float* __restrict__ d,
            const _Float16* __restrict__ whPack, float* __restrict__ out) {
    __shared__ float redAcc[16 * F_];     // [M][f] numerator accumulator (8 KB)
    __shared__ float redL[W_ * 32];       // per (wave,lane) partial row sums
    __shared__ float rsArr[16];           // per-row softmax denominators

    int tidb = threadIdx.x;               // 0..255
    int w = tidb >> 5, lane = tidb & 31, hi = lane >> 4, col = lane & 15;
    int blk = blockIdx.x;                 // i-tile id: 0..511
    int b = blk >> 6, it = blk & 63;
    int i = it * 16 + col;                // A-fragment row for this lane

    // zero the LDS accumulator
    for (int k = tidb; k < 16 * F_; k += 32 * W_) redAcc[k] = 0.f;

    float si = s[b * N_ + i];
    float ae[E_];
#pragma unroll
    for (int e = 0; e < E_; ++e) ae[e] = a_edge[e];

    const float* erow = edges + (size_t)(b * N_ + i) * N_ * E_;
    const float* dv   = d + b * N_;

    v8f acc[8] = {};
    float lsum = 0.f;

    __syncthreads();                      // redAcc zeroed before atomics

#pragma unroll
    for (int m = 0; m < (N_ / 32) / W_; ++m) {
        int kt = w + m * W_;
        if (m + 1 < (N_ / 32) / W_)
            __builtin_prefetch(erow + (size_t)(kt + W_) * 32 * E_, 0, 1);

        // preload all 8 B fragments so loads clause/overlap with score math
        const v16h* bbase = (const v16h*)whPack + (size_t)((b * 32 + kt) * 8) * 32 + lane;
        v16h bfr[8];
#pragma unroll
        for (int ft = 0; ft < 8; ++ft) bfr[ft] = bbase[ft * 32];

        v16h afrag;
#pragma unroll
        for (int e = 0; e < 16; ++e) {
            int j = kt * 32 + frag_k(e, hi);
            const v4f* ep = (const v4f*)(erow + (size_t)j * E_);
            v4f e0 = __builtin_nontemporal_load(ep + 0);
            v4f e1 = __builtin_nontemporal_load(ep + 1);
            v4f e2 = __builtin_nontemporal_load(ep + 2);
            v4f e3 = __builtin_nontemporal_load(ep + 3);
            float dot = e0.x * ae[0]  + e0.y * ae[1]  + e0.z * ae[2]  + e0.w * ae[3]
                      + e1.x * ae[4]  + e1.y * ae[5]  + e1.z * ae[6]  + e1.w * ae[7]
                      + e2.x * ae[8]  + e2.y * ae[9]  + e2.z * ae[10] + e2.w * ae[11]
                      + e3.x * ae[12] + e3.y * ae[13] + e3.z * ae[14] + e3.w * ae[15];
            float t = tanhf(si + dv[j] + dot);
            float wgt = __expf(t);        // t in (-1,1): overflow impossible
            lsum += wgt;
            afrag[e] = (_Float16)wgt;
        }
#pragma unroll
        for (int ft = 0; ft < 8; ++ft)
            acc[ft] = __builtin_amdgcn_wmma_f32_16x16x32_f16(
                false, afrag, false, bfr[ft], (short)0, acc[ft], false, false);
    }

    // combine partial numerators across waves: D-fragment lane (L) holds
    // rows M = r + (hi?8:0), col = ft*16 + (L%16)
    int base = hi ? 8 : 0;
#pragma unroll
    for (int ft = 0; ft < 8; ++ft)
#pragma unroll
        for (int r = 0; r < 8; ++r)
            atomicAdd(&redAcc[(base + r) * F_ + ft * 16 + col], acc[ft][r]);
    redL[w * 32 + lane] = lsum;
    __syncthreads();

    // per-row softmax denominators: lanes (L, L+16) of each wave jointly cover a row
    if (tidb < 16) {
        float rs = 0.f;
#pragma unroll
        for (int ww = 0; ww < W_; ++ww)
            rs += redL[ww * 32 + tidb] + redL[ww * 32 + tidb + 16];
        rsArr[tidb] = rs;
    }
    __syncthreads();

    // finalize: 2048 outputs / 256 threads = 8 each, fully coalesced
    float* otile = out + (size_t)(b * N_ + it * 16) * F_;
    int f = tidb & (F_ - 1);
    int mh = tidb >> 7;                   // 0 or 1
#pragma unroll
    for (int q = 0; q < 8; ++q) {
        int M = 2 * q + mh;
        otile[(size_t)M * F_ + f] = tanhf(redAcc[M * F_ + f] / rsArr[M]);
    }
}

// ---------------------------------------------------------------------------
extern "C" void kernel_launch(void* const* d_in, const int* in_sizes, int n_in,
                              void* d_out, int out_size, void* d_ws, size_t ws_size,
                              hipStream_t stream) {
    (void)in_sizes; (void)n_in; (void)out_size; (void)ws_size;

    const float* h      = (const float*)d_in[0];
    const float* edges  = (const float*)d_in[1];
    const float* Wa     = (const float*)d_in[2];
    const float* a_src  = (const float*)d_in[3];
    const float* a_dst  = (const float*)d_in[4];
    const float* a_edge = (const float*)d_in[5];
    float* out = (float*)d_out;

    // workspace carve-up (~6.2 MB total)
    char* ws = (char*)d_ws;
    float*    Wh     = (float*)   (ws);                            // 8192*128*4 = 4 MB
    _Float16* whPack = (_Float16*)(ws + (4u << 20));               // 8192*128*2 = 2 MB
    _Float16* waPack = (_Float16*)(ws + (6u << 20));               // 128*128*2  = 32 KB
    float*    sv     = (float*)   (ws + (6u << 20) + (64u << 10)); // 8192*4 = 32 KB
    float*    dv     = (float*)   (ws + (6u << 20) + (96u << 10)); // 8192*4 = 32 KB

    pack_wa_kernel<<<8,    128, 0, stream>>>(Wa, waPack);               // 32 waves
    wh_kernel     <<<128,  128, 0, stream>>>(h, waPack, Wh);            // 512 waves
    sd_kernel     <<<1024, 256, 0, stream>>>(Wh, a_src, a_dst, sv, dv); // 8192 waves
    pack_wh_kernel<<<256,  256, 0, stream>>>(Wh, whPack);               // 2048 waves
    attn_kernel   <<<512,  32 * W_, 0, stream>>>(edges, a_edge, sv, dv, whPack, out); // 4096 waves
}
